// LSTMSequenceModel_58531814310445
// MI455X (gfx1250) — matrix operand order repaired and checked by reading, not compiled, not run on hardware
//
#include <hip/hip_runtime.h>
#include <hip/hip_bf16.h>

// ---- sizes (fixed by the reference) ----
#define BATCH 1024
#define TSEQ  512
#define FDIM  128
#define HDIM  8
#define GDIM  32              // 4*H
#define MROWS (BATCH * TSEQ)  // 524288

typedef _Float16 v16h __attribute__((ext_vector_type(16)));
typedef float    v8f  __attribute__((ext_vector_type(8)));
typedef float    v2f  __attribute__((ext_vector_type(2)));

__device__ __forceinline__ float fsigmoid(float x) {
    return 1.0f / (1.0f + __expf(-x));
}

// =====================================================================
// Kernel 1: gx[T, B, 32] = x @ W_ih^T + (b_ih + b_hh)
// One wave = 16(M) x 32(N) tile, K=128 in 4 chunks of 32 via
// v_wmma_f32_16x16x32_f16. Memory-bound: streams x (256 MB) once.
// Output layout [T, B, 32] makes the scan's per-step reads contiguous.
// =====================================================================
__global__ __launch_bounds__(256) void gx_gemm_kernel(
        const float* __restrict__ x,     // [B, T, F] row r = b*T + t
        const float* __restrict__ W_ih,  // [32, 128]
        const float* __restrict__ b_ih,  // [32]
        const float* __restrict__ b_hh,  // [32]
        float* __restrict__ gx)          // [T, B, 32]
{
    const int lane = threadIdx.x & 31;
    const int wave = threadIdx.x >> 5;
    const int half = lane >> 4;          // K-half selector
    const int l16  = lane & 15;
    const int mbase = (blockIdx.x * 8 + wave) * 16;   // row-tile base

    // --- B fragments (W_ih^T, 32x16 f16 tiles): 4 k-chunks x 2 n-tiles ---
    // lane l: N = l%16; idx<8 -> K = kchunk+idx+8*(l/16), idx>=8 -> K = kchunk+idx+8+8*(l/16)
    v16h bfrag[4][2];
    #pragma unroll
    for (int kc = 0; kc < 4; ++kc) {
        #pragma unroll
        for (int nt = 0; nt < 2; ++nt) {
            const int g = nt * 16 + l16;
            const float* wrow = W_ih + g * FDIM + kc * 32 + half * 8;
            v16h b;
            #pragma unroll
            for (int i = 0; i < 16; ++i) b[i] = (_Float16)wrow[i < 8 ? i : i + 8];
            bfrag[kc][nt] = b;
        }
    }

    // --- A fragments + MACs ---
    // A 16x32 f16 layout: lane l: M = l%16; idx<8 -> K = kc+idx+8*(l/16),
    // idx>=8 -> K = kc+idx+8+8*(l/16)
    v8f acc0 = {}; v8f acc1 = {};
    const float* xrow = x + (size_t)(mbase + l16) * FDIM + half * 8;
    #pragma unroll
    for (int kc = 0; kc < 4; ++kc) {
        v16h a;
        #pragma unroll
        for (int i = 0; i < 8; ++i) a[i]     = (_Float16)xrow[kc * 32 + i];
        #pragma unroll
        for (int i = 0; i < 8; ++i) a[8 + i] = (_Float16)xrow[kc * 32 + 16 + i];
        acc0 = __builtin_amdgcn_wmma_f32_16x16x32_f16(false, a, false, bfrag[kc][0],
                                                      (short)0, acc0, false, false);
        acc1 = __builtin_amdgcn_wmma_f32_16x16x32_f16(false, a, false, bfrag[kc][1],
                                                      (short)0, acc1, false, false);
    }

    // --- bias + store in [T, B, 32]; all 16 rows of a tile share b (T%16==0) ---
    const int bidx = mbase / TSEQ;
    const int t0   = mbase % TSEQ;
    const float bias0 = b_ih[l16]      + b_hh[l16];
    const float bias1 = b_ih[16 + l16] + b_hh[16 + l16];
    #pragma unroll
    for (int v = 0; v < 8; ++v) {
        const int t = t0 + v + 8 * half;            // D layout: M = v + 8*(l/16)
        float* dst = gx + ((size_t)t * BATCH + bidx) * GDIM;
        dst[l16]      = acc0[v] + bias0;
        dst[16 + l16] = acc1[v] + bias1;
    }
}

// =====================================================================
// Kernel 2: sequential LSTM scan + decoder.
// One wave per 16 batch rows (64 waves). Per step:
//   acc(init from gx) += h @ W_hh^T  via 4x v_wmma_f32_16x16x4_f32
//   gates -> c,h update (one v_exp per sigmoid/tanh, shfl_xor for i/f g/o)
// h transposed D-layout -> A-layout through 512 B of LDS.
// =====================================================================
__global__ __launch_bounds__(32) void lstm_scan_kernel(
        const float* __restrict__ gx,     // [T, B, 32]
        const float* __restrict__ W_hh,   // [32, 8]
        const float* __restrict__ W_dec,  // [128, 8]
        const float* __restrict__ b_dec,  // [128]
        float* __restrict__ out,          // [B, 128]
        float* __restrict__ h_out,        // [B, 8]
        float* __restrict__ c_out)        // [B, 8]
{
    __shared__ float hlds[16 * HDIM];     // 16 batch rows x 8 h values

    const int lane  = threadIdx.x & 31;
    const int half  = lane >> 4;
    const int n     = lane & 15;          // column within 16-wide N tile
    const int bbase = blockIdx.x * 16;

    // --- W_hh^T B fragments (4x16 f32): kc in {0,4}, nt in {0,1}; live in
    //     registers for all 512 steps.
    //     lane l: N = l%16; V0/V1 = K = kc + 2*(l/16) + {0,1}
    v2f bf[2][2];
    #pragma unroll
    for (int kci = 0; kci < 2; ++kci) {
        #pragma unroll
        for (int nt = 0; nt < 2; ++nt) {
            const int g = nt * 16 + n;
            const int k = kci * 4 + 2 * half;
            v2f b; b.x = W_hh[g * HDIM + k]; b.y = W_hh[g * HDIM + k + 1];
            bf[kci][nt] = b;
        }
    }

    // h,c state in D-layout: lanes with n<8 hold column j=n of rows m=v+8*half
    float hreg[8], creg[8];
    #pragma unroll
    for (int v = 0; v < 8; ++v) { hreg[v] = 0.0f; creg[v] = 0.0f; }

    for (int t = 0; t < TSEQ; ++t) {
        // load gx for this step directly into the C/D accumulator layout;
        // [T,B,32] layout => the wave reads 2 KB contiguous
        const float* grow = gx + ((size_t)t * BATCH + bbase) * GDIM;
        v8f acc0, acc1;
        #pragma unroll
        for (int v = 0; v < 8; ++v) {
            const int m = v + 8 * half;
            acc0[v] = grow[m * GDIM + n];
            acc1[v] = grow[m * GDIM + 16 + n];
        }
        // prefetch next step's slab (global_prefetch_b8) — independent of h
        if (t + 1 < TSEQ)
            __builtin_prefetch(gx + ((size_t)(t + 1) * BATCH + bbase) * GDIM + lane * 16, 0, 0);

        // transpose h (D-layout) -> A-layout via LDS
        if (n < HDIM) {
            #pragma unroll
            for (int v = 0; v < 8; ++v)
                hlds[(v + 8 * half) * HDIM + n] = hreg[v];
        }
        __syncthreads();
        // A 16x4 f32 layout: lane l: M = l%16, V0/V1 = K = kc + 2*(l/16) + {0,1}
        v2f a0, a1;
        a0.x = hlds[n * HDIM + 2 * half];     a0.y = hlds[n * HDIM + 2 * half + 1];
        a1.x = hlds[n * HDIM + 4 + 2 * half]; a1.y = hlds[n * HDIM + 4 + 2 * half + 1];
        __syncthreads();

        acc0 = __builtin_amdgcn_wmma_f32_16x16x4_f32(false, a0, false, bf[0][0],
                                                     (short)0, acc0, false, false);
        acc0 = __builtin_amdgcn_wmma_f32_16x16x4_f32(false, a1, false, bf[1][0],
                                                     (short)0, acc0, false, false);
        acc1 = __builtin_amdgcn_wmma_f32_16x16x4_f32(false, a0, false, bf[0][1],
                                                     (short)0, acc1, false, false);
        acc1 = __builtin_amdgcn_wmma_f32_16x16x4_f32(false, a1, false, bf[1][1],
                                                     (short)0, acc1, false, false);

        // gates: acc0 = [i | f], acc1 = [g | o] across the 16 N columns.
        #pragma unroll
        for (int v = 0; v < 8; ++v) {
            const float s0 = fsigmoid(acc0[v]);            // i (n<8) or f (n>=8)
            // one exp for tanh-or-sigmoid: tanh(x) = 2*sigmoid(2x) - 1
            const float xs = (n < HDIM) ? 2.0f * acc1[v] : acc1[v];
            const float r  = 1.0f / (1.0f + __expf(-xs));
            const float s1 = (n < HDIM) ? (2.0f * r - 1.0f) : r;  // g or o
            const float fv = __shfl_xor(s0, 8, 32);        // partner: f for n<8
            const float ov = __shfl_xor(s1, 8, 32);        // partner: o for n<8
            const float c  = fv * creg[v] + s0 * s1;       // valid on lanes n<8
            creg[v] = c;
            const float e2c = __expf(-2.0f * c);
            hreg[v] = ov * (2.0f / (1.0f + e2c) - 1.0f);   // o * tanh(c)
        }
    }

    // --- final h, c outputs + stage h for the decoder ---
    if (n < HDIM) {
        #pragma unroll
        for (int v = 0; v < 8; ++v) {
            const int b = bbase + v + 8 * half;
            h_out[b * HDIM + n] = hreg[v];
            c_out[b * HDIM + n] = creg[v];
            hlds[(v + 8 * half) * HDIM + n] = hreg[v];
        }
    }
    __syncthreads();

    // --- decoder: out[b][f] = h[b] . W_dec[f] + b_dec[f] (tiny; scalar) ---
    #pragma unroll
    for (int f0 = 0; f0 < 4; ++f0) {
        const int f = f0 * 32 + lane;
        float wd[HDIM];
        #pragma unroll
        for (int j = 0; j < HDIM; ++j) wd[j] = W_dec[f * HDIM + j];
        const float bd = b_dec[f];
        for (int m = 0; m < 16; ++m) {
            float s = bd;
            #pragma unroll
            for (int j = 0; j < HDIM; ++j) s += hlds[m * HDIM + j] * wd[j];
            out[(size_t)(bbase + m) * FDIM + f] = s;   // lanes contiguous in f
        }
    }
}

extern "C" void kernel_launch(void* const* d_in, const int* in_sizes, int n_in,
                              void* d_out, int out_size, void* d_ws, size_t ws_size,
                              hipStream_t stream) {
    const float* x     = (const float*)d_in[0];  // [B,T,F]
    const float* W_ih  = (const float*)d_in[1];  // [32,128]
    const float* W_hh  = (const float*)d_in[2];  // [32,8]
    const float* b_ih  = (const float*)d_in[3];  // [32]
    const float* b_hh  = (const float*)d_in[4];  // [32]
    const float* W_dec = (const float*)d_in[5];  // [128,8]
    const float* b_dec = (const float*)d_in[6];  // [128]

    float* out   = (float*)d_out;                 // [B,128]
    float* h_out = out + (size_t)BATCH * FDIM;    // [1,B,8]
    float* c_out = h_out + (size_t)BATCH * HDIM;  // [1,B,8]
    float* gx    = (float*)d_ws;                  // [T,B,32] = 64 MB scratch

    // Phase 1: 524288 rows / (8 waves * 16 rows) = 4096 blocks
    gx_gemm_kernel<<<4096, 256, 0, stream>>>(x, W_ih, b_ih, b_hh, gx);
    // Phase 2: 64 waves, one per 16 batch rows
    lstm_scan_kernel<<<BATCH / 16, 32, 0, stream>>>(gx, W_hh, W_dec, b_dec,
                                                    out, h_out, c_out);
}
